// MAEGIN_9749575762317
// MI455X (gfx1250) — matrix operand order repaired and checked by reading, not compiled, not run on hardware
//
#include <hip/hip_runtime.h>

#define N_NODES 50000
#define N_EDGES 800000
#define D_DIM   256
#define VOCABSZ 4096
#define BN_EPS  1e-5f

typedef __attribute__((ext_vector_type(16))) _Float16 v16h;
typedef __attribute__((ext_vector_type(8)))  _Float16 v8h;
typedef __attribute__((ext_vector_type(8)))  float    v8f;
typedef __attribute__((ext_vector_type(4)))  float    v4f;

// -------- embedding gather: h[i,:] = emb[x[i],:] (float4 vectorized) --------
__global__ void gather_embed_kernel(const int* __restrict__ x,
                                    const float* __restrict__ emb,
                                    float* __restrict__ h) {
  int i = blockIdx.x * blockDim.x + threadIdx.x;  // index of one float4
  int node = i >> 6;                              // 64 float4 per node (D=256)
  int c4 = i & 63;
  v4f v = ((const v4f*)(emb + (size_t)x[node] * D_DIM))[c4];
  ((v4f*)(h + (size_t)node * D_DIM))[c4] = v;
}

// -------- plain float4 copy (self term of GIN aggregation) --------
__global__ void copy_kernel(const float* __restrict__ in, float* __restrict__ out) {
  size_t i = (size_t)blockIdx.x * blockDim.x + threadIdx.x;
  ((v4f*)out)[i] = ((const v4f*)in)[i];
}

// -------- edge scatter: out[dst] += h[src] via f32 global atomics --------
// thread t handles 8 contiguous floats of one edge; a wave covers one full
// 1KB feature row -> coalesced reads; destination (51MB) is L2-resident.
__global__ void edge_scatter_kernel(const int* __restrict__ src,
                                    const int* __restrict__ dst,
                                    const float* __restrict__ h,
                                    float* __restrict__ out) {
  size_t t = (size_t)blockIdx.x * blockDim.x + threadIdx.x;
  int e = (int)(t >> 5);
  int seg = ((int)t & 31) * 8;
  int s = src[e], d = dst[e];
  const float* hp = h + (size_t)s * D_DIM + seg;
  float* op = out + (size_t)d * D_DIM + seg;
#pragma unroll
  for (int j = 0; j < 8; ++j) {
    (void)__hip_atomic_fetch_add(op + j, hp[j], __ATOMIC_RELAXED,
                                 __HIP_MEMORY_SCOPE_AGENT);
  }
}

// -------- fused WMMA linear:  out = epi(A[M,256] @ W[N,256]^T) --------
// epi = (+bias) -> (BN eval, PReLU) -> (+residual), all optional via nullptr.
// Block = 256 threads = 8 waves; block tile 16(M) x 128(N); wave tile 16x16.
// A tile staged fp32 in LDS via gfx1250 async global->LDS DMA (ASYNCcnt),
// row pad +4 floats -> row stride 1040B -> conflict-free ds_load_b128 frags.
__launch_bounds__(256)
__global__ void wmma_linear_kernel(const float* __restrict__ A,
                                   const float* __restrict__ W,
                                   const float* __restrict__ bias,
                                   const float* __restrict__ gamma,
                                   const float* __restrict__ beta,
                                   const float* __restrict__ mean,
                                   const float* __restrict__ var,
                                   const float* __restrict__ alpha,
                                   const float* __restrict__ residual,
                                   float* __restrict__ out, int N) {
  __shared__ float Asf[16][D_DIM + 4];   // 16 x 260 f32, row stride 1040B

  const int tid = threadIdx.x;
  const int mbase = blockIdx.y * 16;

  // ---- async-stage A tile (16 x 256 fp32 = 16KB) straight into LDS ----
  // 1024 16-byte chunks; each of 256 threads issues 4 async b128 copies.
  {
    const uint32_t lds0 = (uint32_t)(uintptr_t)&Asf[0][0];
#pragma unroll
    for (int it = 0; it < 4; ++it) {
      int c = it * 256 + tid;            // chunk id
      int row = c >> 6;                  // 64 chunks per row
      int col = (c & 63) << 2;           // float column
      uint64_t ga = (uint64_t)(uintptr_t)(A + (size_t)(mbase + row) * D_DIM + col);
      uint32_t la = lds0 + (uint32_t)((row * (D_DIM + 4) + col) * 4);
      asm volatile("global_load_async_to_lds_b128 %0, %1, off"
                   :: "v"(la), "v"(ga) : "memory");
    }
    asm volatile("s_wait_asynccnt 0" ::: "memory");
  }
  __syncthreads();

  const int wave = tid >> 5;
  const int lane = tid & 31;
  const int nbase = blockIdx.x * 128 + wave * 16;
  const int lrow = lane & 15;
  const int khalfA = (lane & 16) ? 8 : 0;    // A layout: K {kb..kb+7, kb+16..kb+23}
  const int kbB   = (lane & 16) ? 8 : 0;     // B layout: K {kb..kb+7, kb+16..kb+23}
  const int n = nbase + lrow;                // B column n == row n of W
  const float* wrow = W + (size_t)n * D_DIM;

  v8f acc = {};
#pragma unroll
  for (int k0 = 0; k0 < D_DIM; k0 += 32) {
    // A fragment from LDS (fp32 -> f16 at fragment build; cvt co-executes)
    const int ka = k0 + khalfA;
    const v4f* ap0 = (const v4f*)&Asf[lrow][ka];
    v4f a0 = ap0[0];
    v4f a1 = ap0[1];
    const v4f* ap1 = (const v4f*)&Asf[lrow][ka + 16];
    v4f a2 = ap1[0];
    v4f a3 = ap1[1];
    v16h a;
#pragma unroll
    for (int j = 0; j < 4; ++j) {
      a[j]      = (_Float16)a0[j];
      a[4 + j]  = (_Float16)a1[j];
      a[8 + j]  = (_Float16)a2[j];
      a[12 + j] = (_Float16)a3[j];
    }

    // B fragment from global (W is L2-resident); prefetch next K-slab near
    const int kb = k0 + kbB;
    __builtin_prefetch(wrow + k0 + 32, 0, 3);
    v4f w0 = *(const v4f*)(wrow + kb);
    v4f w1 = *(const v4f*)(wrow + kb + 4);
    v4f w2 = *(const v4f*)(wrow + kb + 16);
    v4f w3 = *(const v4f*)(wrow + kb + 20);
    v16h b;
#pragma unroll
    for (int j = 0; j < 4; ++j) {
      b[j]      = (_Float16)w0[j];
      b[4 + j]  = (_Float16)w1[j];
      b[8 + j]  = (_Float16)w2[j];
      b[12 + j] = (_Float16)w3[j];
    }

    acc = __builtin_amdgcn_wmma_f32_16x16x32_f16(
        /*neg_a=*/false, a, /*neg_b=*/false, b,
        /*c_mod=*/(short)0, acc, /*reuse_a=*/false, /*reuse_b=*/false);
  }

  // ---- fused epilogue ----
  const bool has_bn = (gamma != nullptr);
  float bv = bias ? bias[n] : 0.0f;
  float bn_s = 1.0f, bn_o = 0.0f, al = 0.0f;
  if (has_bn) {
    float g = gamma[n], be = beta[n], mu = mean[n], vv = var[n];
    float rs = rsqrtf(vv + BN_EPS);
    bn_s = g * rs;
    bn_o = be - mu * g * rs;
    al = *alpha;
  }
  const int mlane = mbase + ((lane >> 4) << 3);  // C/D layout: lanes16-31 -> M+8
#pragma unroll
  for (int r = 0; r < 8; ++r) {
    int m = mlane + r;
    float v = acc[r] + bv;
    if (has_bn) {
      v = v * bn_s + bn_o;
      v = (v >= 0.0f) ? v : al * v;   // PReLU
    }
    if (residual) v += residual[(size_t)m * N + n];
    out[(size_t)m * N + n] = v;
  }
}

static void launch_mm(const float* A, const float* W, const float* bias,
                      const float* gamma, const float* beta, const float* mean,
                      const float* var, const float* alpha, const float* residual,
                      float* out, int N, hipStream_t stream) {
  dim3 grid(N / 128, N_NODES / 16);   // 50000/16 = 3125 exactly, no tails
  hipLaunchKernelGGL(wmma_linear_kernel, grid, dim3(256), 0, stream,
                     A, W, bias, gamma, beta, mean, var, alpha, residual, out, N);
}

extern "C" void kernel_launch(void* const* d_in, const int* in_sizes, int n_in,
                              void* d_out, int out_size, void* d_ws, size_t ws_size,
                              hipStream_t stream) {
  (void)in_sizes; (void)n_in; (void)out_size; (void)ws_size;

  const int*   x     = (const int*)d_in[0];
  const int*   ei    = (const int*)d_in[1];
  const int*   src   = ei;
  const int*   dst   = ei + N_EDGES;
  const float* emb   = (const float*)d_in[2];
  const float* W1s   = (const float*)d_in[3];
  const float* b1s   = (const float*)d_in[4];
  const float* g1s   = (const float*)d_in[5];
  const float* be1s  = (const float*)d_in[6];
  const float* m1s   = (const float*)d_in[7];
  const float* v1s   = (const float*)d_in[8];
  const float* a1s   = (const float*)d_in[9];
  const float* W2s   = (const float*)d_in[10];
  const float* b2s   = (const float*)d_in[11];
  const float* g2s   = (const float*)d_in[12];
  const float* be2s  = (const float*)d_in[13];
  const float* m2s   = (const float*)d_in[14];
  const float* v2s   = (const float*)d_in[15];
  const float* a2s   = (const float*)d_in[16];
  const float* Wproj = (const float*)d_in[17];
  const float* Wtrn  = (const float*)d_in[18];
  const float* btrn  = (const float*)d_in[19];
  const float* Wprd  = (const float*)d_in[20];
  const float* bprd  = (const float*)d_in[21];

  const size_t NDF = (size_t)N_NODES * D_DIM;   // 12.8M floats = 51.2 MB
  float* H = (float*)d_ws;       // current node features
  float* B = H + NDF;            // scratch / aggregation
  float* C = B + NDF;            // scratch

  const int copyBlocks    = (int)(NDF / 4 / 256);        // 12500
  const int scatterBlocks = (N_EDGES * 32) / 256;        // 100000

  // h = emb[x]
  hipLaunchKernelGGL(gather_embed_kernel, dim3(copyBlocks), dim3(256), 0, stream,
                     x, emb, H);

  for (int i = 0; i < 5; ++i) {
    // B = H + segment_sum(H[src], dst)
    hipLaunchKernelGGL(copy_kernel, dim3(copyBlocks), dim3(256), 0, stream, H, B);
    hipLaunchKernelGGL(edge_scatter_kernel, dim3(scatterBlocks), dim3(256), 0, stream,
                       src, dst, H, B);
    // C = PReLU(BN(B @ W1^T + b1))
    launch_mm(B, W1s + (size_t)i * D_DIM * D_DIM, b1s + i * D_DIM,
              g1s + i * D_DIM, be1s + i * D_DIM, m1s + i * D_DIM, v1s + i * D_DIM,
              a1s + i, nullptr, C, D_DIM, stream);
    // B = PReLU(BN(C @ W2^T + b2)) + H   (residual)
    launch_mm(C, W2s + (size_t)i * D_DIM * D_DIM, b2s + i * D_DIM,
              g2s + i * D_DIM, be2s + i * D_DIM, m2s + i * D_DIM, v2s + i * D_DIM,
              a2s + i, H, B, D_DIM, stream);
    float* t = H; H = B; B = t;  // H <- new features
  }

  // B = H @ Wproj^T  (no bias/BN)
  launch_mm(H, Wproj, nullptr, nullptr, nullptr, nullptr, nullptr, nullptr,
            nullptr, B, D_DIM, stream);
  // C = B + segment_sum(B[src], dst)
  hipLaunchKernelGGL(copy_kernel, dim3(copyBlocks), dim3(256), 0, stream, B, C);
  hipLaunchKernelGGL(edge_scatter_kernel, dim3(scatterBlocks), dim3(256), 0, stream,
                     src, dst, B, C);
  // H = C @ Wtrn^T + btrn
  launch_mm(C, Wtrn, btrn, nullptr, nullptr, nullptr, nullptr, nullptr,
            nullptr, H, D_DIM, stream);
  // logits = H @ Wprd^T + bprd  -> d_out [50000, 4096]
  launch_mm(H, Wprd, bprd, nullptr, nullptr, nullptr, nullptr, nullptr,
            nullptr, (float*)d_out, VOCABSZ, stream);
}
